// ACmix_49658411876423
// MI455X (gfx1250) — matrix operand order repaired and checked by reading, not compiled, not run on hardware
//
#include <hip/hip_runtime.h>

// ---------------- problem constants ----------------
#define BSZ    64
#define CCH    256
#define HH     28
#define WW     28
#define HW     784          // 28*28
#define HEADS  4
#define HD     64
#define BLK    4
#define HALO   4
#define WINS   12           // BLOCK + 2*HALO
#define NB     7            // H / BLOCK
#define NWIN   49
#define NQ     16
#define NKTOT  144          // WIN*WIN
#define RELN   23           // 2*WIN-1
#define BIGNEG (-3.402823466e+38f)

typedef __attribute__((ext_vector_type(16))) __bf16 v16bf;
typedef __attribute__((ext_vector_type(8)))  float  v8f;
typedef __attribute__((ext_vector_type(4)))  unsigned v4u;
typedef __attribute__((ext_vector_type(8)))  unsigned v8u;

union FragU {
    v16bf v;
    unsigned int u[8];
};

__device__ __forceinline__ unsigned short f2bf(float f) {
    unsigned int u = __float_as_uint(f);
    u = (u + 0x7FFFu + ((u >> 16) & 1u)) >> 16;   // round-to-nearest-even
    return (unsigned short)u;
}

// A fragment: 16xK tile, element (m,k) in LDS as [m][k] row-major (pairs contiguous).
__device__ __forceinline__ v16bf load_a_frag(const unsigned short* lds, int rowStride,
                                             int kBase, int lane) {
    FragU f;
    int m  = lane & 15;
    int kb = kBase + ((lane >> 4) << 4);
    const unsigned short* p = lds + m * rowStride + kb;
#pragma unroll
    for (int j = 0; j < 4; ++j) {
        f.u[j]     = *(const unsigned int*)(p + 2 * j);
        f.u[4 + j] = *(const unsigned int*)(p + 8 + 2 * j);
    }
    return f.v;
}

// B fragment: Kx16 tile, element (k,n), stored in LDS as [n][k] row-major.
__device__ __forceinline__ v16bf load_b_frag(const unsigned short* lds, int rowStride,
                                             int nBase, int kBase, int lane) {
    FragU f;
    int n  = nBase + (lane & 15);
    int kb = kBase + ((lane >> 4) << 4);
    const unsigned short* p = lds + n * rowStride + kb;
#pragma unroll
    for (int j = 0; j < 8; ++j)
        f.u[j] = *(const unsigned int*)(p + 2 * j);
    return f.v;
}

__device__ __forceinline__ v8f wmma_bf16(v16bf a, v16bf b, v8f c) {
    return __builtin_amdgcn_wmma_f32_16x16x32_bf16(false, a, false, b, (short)0, c,
                                                   false, false);
}

// async copy of 16 bytes  global -> LDS  (ASYNCcnt tracked)
__device__ __forceinline__ void async_b128(unsigned ldsOff, unsigned long long gaddr) {
    asm volatile("global_load_async_to_lds_b128 %0, %1, off"
                 :: "v"(ldsOff), "v"(gaddr) : "memory");
}
__device__ __forceinline__ void wait_async0() {
    asm volatile("s_wait_asynccnt 0x0" ::: "memory");
}

// Tensor Data Mover: load a 2D bf16 tile (tileRows x 32 elems, tensor row stride
// 256 elems) from global to LDS (packed row-major).  D# per ISA 08 sec 8.3/8.4.
// Issued once per wave (EXEC ignored); tracked with TENSORcnt.
__device__ __forceinline__ void tdm_load_tile(unsigned ldsAddr,
                                              unsigned long long gaddr,
                                              unsigned tileRows) {
    v4u g0;
    g0[0] = 1u;                                   // count=1 valid descriptor
    g0[1] = ldsAddr;                              // lds_addr (bytes)
    g0[2] = (unsigned)gaddr;                      // global_addr[31:0]
    g0[3] = (unsigned)(gaddr >> 32) | 0x80000000u;// global_addr[56:32] | type=2
    v8u g1;
    g1[0] = 0x00010000u;                          // data_size=1 (2 bytes)
    g1[1] = 256u << 16;                           // tensor_dim0 = 256
    g1[2] = 256u << 16;                           // tensor_dim1 = 256
    g1[3] = 32u << 16;                            // tile_dim0 = 32 elems
    g1[4] = tileRows;                             // tile_dim1 (tile_dim2 = 0)
    g1[5] = 256u;                                 // tensor_dim0_stride = 256
    g1[6] = 0u;
    g1[7] = 0u;
    asm volatile("tensor_load_to_lds %0, %1" :: "s"(g0), "s"(g1) : "memory");
}

// =======================================================================
// Kernel 0a: w(o,c) fp32 -> bf16  (layout preserved: [mat][o][c])
// =======================================================================
__global__ __launch_bounds__(256) void w_to_bf16(
    const float* __restrict__ w1, const float* __restrict__ w2,
    const float* __restrict__ w3, unsigned short* __restrict__ wbf) {
    int gid = blockIdx.x * 256 + threadIdx.x;          // 3*256*256
    int mat = gid >> 16;
    int idx = gid & 65535;
    const float* w = (mat == 0) ? w1 : (mat == 1) ? w2 : w3;
    wbf[gid] = f2bf(w[idx]);
}

// =======================================================================
// Kernel 0b: x[b][c][p] fp32 -> xT[b][p][c] bf16 (pixel-major, fragment order)
// =======================================================================
__global__ __launch_bounds__(256) void x_to_bf16T(
    const float* __restrict__ x, unsigned short* __restrict__ xT) {
    int gid = blockIdx.x * 256 + threadIdx.x;          // BSZ*HW*CCH
    int c = gid & 255;
    int p = (gid >> 8) % HW;
    int b = gid / (HW * CCH);
    xT[gid] = f2bf(x[((size_t)b * CCH + c) * HW + p]);
}

// =======================================================================
// Kernel 1: q/k/v = x @ W^T + b  via bf16 WMMA.
// A tile staged with per-lane ASYNC copies; B tiles staged with 3 TDM
// tensor_load_to_lds ops per K-step (wave 0 issues, TENSORcnt waited).
// One WG: 16 pixels x 128 outs. grid (BSZ, 49, 2), block 256 (8 waves).
// =======================================================================
__global__ __launch_bounds__(256) void qkv_gemm(
    const unsigned short* __restrict__ xT,
    const unsigned short* __restrict__ wbf,
    const float* __restrict__ b1, const float* __restrict__ b2,
    const float* __restrict__ b3,
    float* __restrict__ q, float* __restrict__ k, float* __restrict__ v) {

    __shared__ unsigned short Alds[16 * 32];          // [pixel][ch] stride 32
    __shared__ unsigned short Blds[3 * 128 * 32];     // [mat][out][ch] stride 32

    const int b    = blockIdx.x;
    const int p0   = blockIdx.y * 16;
    const int n0   = blockIdx.z * 128;
    const int tid  = threadIdx.x;
    const int lane = tid & 31;
    const int wave = tid >> 5;

    const unsigned short* xrow = xT + ((size_t)b * HW + p0) * CCH;   // 16 rows of 256

    v8f acc0 = {0, 0, 0, 0, 0, 0, 0, 0};
    v8f acc1 = acc0, acc2 = acc0;

    for (int ks = 0; ks < 8; ++ks) {
        const int k0 = ks * 32;
        // ---- A tile: 16 rows x 64B via per-lane async copies ----
        if (tid < 64) {
            int row = tid >> 2, part = tid & 3;
            unsigned long long ga =
                (unsigned long long)(size_t)(xrow + (size_t)row * CCH + k0)
                + (unsigned)part * 16u;
            unsigned ldsOff =
                (unsigned)(size_t)(Alds + row * 32) + (unsigned)part * 16u;
            async_b128(ldsOff, ga);
        }
        // ---- B tiles: 3 x (128 x 32) bf16 tiles via Tensor Data Mover ----
        if (tid < 32) {
            unsigned bbase = (unsigned)(size_t)Blds;
#pragma unroll
            for (int mat = 0; mat < 3; ++mat) {
                unsigned long long ga = (unsigned long long)(size_t)
                    (wbf + (size_t)mat * 65536 + (size_t)n0 * CCH + k0);
                tdm_load_tile(bbase + (unsigned)mat * 8192u, ga, 128u);
            }
            __builtin_amdgcn_s_wait_tensorcnt(0);   // wave0's TDM ops done
        }
        if (ks < 7)                                 // global_prefetch_b8 hint
            __builtin_prefetch(xrow + (size_t)(tid & 15) * CCH + k0 + 32, 0, 0);

        wait_async0();            // wave's async copies landed in LDS
        __syncthreads();          // all staging visible to all waves

        v16bf a   = load_a_frag(Alds, 32, 0, lane);
        v16bf bb0 = load_b_frag(Blds,        32, wave * 16, 0, lane);
        v16bf bb1 = load_b_frag(Blds + 4096, 32, wave * 16, 0, lane);
        v16bf bb2 = load_b_frag(Blds + 8192, 32, wave * 16, 0, lane);
        acc0 = wmma_bf16(a, bb0, acc0);
        acc1 = wmma_bf16(a, bb1, acc1);
        acc2 = wmma_bf16(a, bb2, acc2);
        __syncthreads();          // fragment reads done before next staging round
    }

    const int n    = n0 + wave * 16 + (lane & 15);
    const int mofs = (lane >> 4) << 3;
    const float bq = b1[n], bk = b2[n], bv = b3[n];
    const size_t obase = ((size_t)b * CCH + n) * HW + p0 + mofs;

    float4 t;
    t = make_float4(acc0[0] + bq, acc0[1] + bq, acc0[2] + bq, acc0[3] + bq);
    *(float4*)(q + obase) = t;
    t = make_float4(acc0[4] + bq, acc0[5] + bq, acc0[6] + bq, acc0[7] + bq);
    *(float4*)(q + obase + 4) = t;
    t = make_float4(acc1[0] + bk, acc1[1] + bk, acc1[2] + bk, acc1[3] + bk);
    *(float4*)(k + obase) = t;
    t = make_float4(acc1[4] + bk, acc1[5] + bk, acc1[6] + bk, acc1[7] + bk);
    *(float4*)(k + obase + 4) = t;
    t = make_float4(acc2[0] + bv, acc2[1] + bv, acc2[2] + bv, acc2[3] + bv);
    *(float4*)(v + obase) = t;
    t = make_float4(acc2[4] + bv, acc2[5] + bv, acc2[6] + bv, acc2[7] + bv);
    *(float4*)(v + obase + 4) = t;
}

// =======================================================================
// Kernel 2: halo attention.  grid (BSZ, NWIN), block 128 (wave == head).
// LDS arena (62976 B):
//   [0      .. 8448 )  q tile   16x264  bf16      (phase A..C)
//   [8448   .. 16896)  k chunk  16x264  bf16      (phase B)
//   [16896  .. 25088)  rel      2x32x64 bf16      (phase C)
//   [25088  .. 62976)  sim      4x16x148 f32      (phase B..D)
// overlay after softmax:
//   [0      .. 20480)  attn     4x16x160 bf16
//   [20480  .. 37888)  v chunk  256x34   bf16
// =======================================================================
__global__ __launch_bounds__(128) void halo_attn(
    const float* __restrict__ qg, const float* __restrict__ kg,
    const float* __restrict__ vg,
    const float* __restrict__ relh, const float* __restrict__ relw,
    const float* __restrict__ rate1p, float* __restrict__ out) {

    __shared__ __align__(16) char smem[62976];
    unsigned short* qlds    = (unsigned short*)(smem);
    unsigned short* klds    = (unsigned short*)(smem + 8448);
    unsigned short* rlds    = (unsigned short*)(smem + 16896);
    float*          simlds  = (float*)(smem + 25088);
    unsigned short* attnlds = (unsigned short*)(smem);
    unsigned short* vlds    = (unsigned short*)(smem + 20480);

    const int b    = blockIdx.x;
    const int win  = blockIdx.y;
    const int y0   = (win / NB) * BLK;
    const int x0   = (win % NB) * BLK;
    const int tid  = threadIdx.x;
    const int lane = tid & 31;
    const int h    = tid >> 5;               // head == wave

    const size_t base = (size_t)b * CCH * HW;
    const float* qb = qg + base;
    const float* kb = kg + base;
    const float* vb = vg + base;
    const float scaling = 0.125f;            // HD^-0.5

    // ---- phase A: stage q block (scaled) and rel tables ----
#pragma unroll
    for (int i = 0; i < 32; ++i) {
        int idx = tid + i * 128;             // 16*256
        int qp = idx >> 8, ch = idx & 255;
        int gp = (y0 + (qp >> 2)) * WW + x0 + (qp & 3);
        qlds[qp * 264 + ch] = f2bf(qb[(size_t)ch * HW + gp] * scaling);
    }
#pragma unroll
    for (int i = 0; i < 32; ++i) {
        int idx = tid + i * 128;             // 2*32*64
        int t = idx >> 11, r = (idx >> 6) & 31, d = idx & 63;
        float val = (r < RELN) ? (t == 0 ? relw[r * HD + d] : relh[r * HD + d]) : 0.f;
        rlds[t * 2048 + r * HD + d] = f2bf(val);
    }
    __syncthreads();

    // q fragments for this head (reused in QK^T and rel-pos)
    v16bf qa0 = load_a_frag(qlds, 264, h * HD + 0,  lane);
    v16bf qa1 = load_a_frag(qlds, 264, h * HD + 32, lane);

    const int Nl    = lane & 15;
    const int mbase = (lane >> 4) << 3;

    // ---- phase B: sim = q k^T over 9 chunks of 16 window positions ----
#pragma unroll
    for (int ci = 0; ci < 9; ++ci) {
#pragma unroll
        for (int i = 0; i < 32; ++i) {
            int idx = tid + i * 128;
            int l = idx >> 8, ch = idx & 255;
            int wp = ci * 16 + l;
            int gy = y0 - HALO + wp / WINS;
            int gx = x0 - HALO + wp % WINS;
            float val = (gy >= 0 && gy < HH && gx >= 0 && gx < WW)
                        ? kb[(size_t)ch * HW + gy * WW + gx] : 0.f;
            klds[l * 264 + ch] = f2bf(val);
        }
        __syncthreads();
        v16bf kf0 = load_b_frag(klds, 264, 0, h * HD + 0,  lane);
        v16bf kf1 = load_b_frag(klds, 264, 0, h * HD + 32, lane);
        v8f s = {0, 0, 0, 0, 0, 0, 0, 0};
        s = wmma_bf16(qa0, kf0, s);
        s = wmma_bf16(qa1, kf1, s);
#pragma unroll
        for (int r = 0; r < 8; ++r)
            simlds[h * 2368 + (mbase + r) * 148 + ci * 16 + Nl] = s[r];
        __syncthreads();
    }

    // ---- phase C: relative position logits (WMMA, scatter into sim) ----
#pragma unroll
    for (int t = 0; t < 2; ++t) {            // t=0 width, t=1 height
#pragma unroll
        for (int nt = 0; nt < 2; ++nt) {
            v16bf rb0 = load_b_frag(rlds + t * 2048, 64, nt * 16, 0,  lane);
            v16bf rb1 = load_b_frag(rlds + t * 2048, 64, nt * 16, 32, lane);
            v8f rs = {0, 0, 0, 0, 0, 0, 0, 0};
            rs = wmma_bf16(qa0, rb0, rs);
            rs = wmma_bf16(qa1, rb1, rs);
            int N = nt * 16 + Nl;
#pragma unroll
            for (int r = 0; r < 8; ++r) {
                int m = mbase + r;
                int qy = m >> 2, qx = m & 3;
                if (N < RELN) {
                    if (t == 0) {
                        int wx = N - (WINS - 1) + qx;
                        if (wx >= 0 && wx < WINS)
                            for (int wy = 0; wy < WINS; ++wy)
                                simlds[h * 2368 + m * 148 + wy * WINS + wx] += rs[r];
                    } else {
                        int wy = N - (WINS - 1) + qy;
                        if (wy >= 0 && wy < WINS)
                            for (int wx = 0; wx < WINS; ++wx)
                                simlds[h * 2368 + m * 148 + wy * WINS + wx] += rs[r];
                    }
                }
            }
        }
    }
    __syncthreads();   // attn overlay will clobber q/k/rel regions

    // ---- phase D: mask (reference's inverted valid mask) + softmax -> bf16 attn ----
    if (lane < 16) {
        int m = lane;
        float* srow = simlds + h * 2368 + m * 148;
        float mx = BIGNEG;
        for (int kk = 0; kk < NKTOT; ++kk) {
            int gy = y0 - HALO + kk / WINS;
            int gx = x0 - HALO + kk % WINS;
            bool inb = (gy >= 0 && gy < HH && gx >= 0 && gx < WW);
            float sv = inb ? BIGNEG : srow[kk];      // reference: valid -> big_neg
            srow[kk] = sv;
            mx = fmaxf(mx, sv);
        }
        float sum = 0.f;
        for (int kk = 0; kk < NKTOT; ++kk) {
            float e = __expf(srow[kk] - mx);
            srow[kk] = e;
            sum += e;
        }
        float inv = 1.f / sum;
        for (int kk = 0; kk < 160; ++kk) {
            float a = (kk < NKTOT) ? srow[kk] * inv : 0.f;
            attnlds[h * 2560 + m * 160 + kk] = f2bf(a);
        }
    }
    __syncthreads();   // sim region becomes v-tile region

    // ---- phase E: out = attn @ v  (K = 160 in 5 chunks of 32) ----
    v8f o[4];
#pragma unroll
    for (int nt = 0; nt < 4; ++nt) o[nt] = (v8f){0, 0, 0, 0, 0, 0, 0, 0};

#pragma unroll
    for (int kc = 0; kc < 5; ++kc) {
#pragma unroll
        for (int i = 0; i < 64; ++i) {
            int idx = tid + i * 128;             // 256*32
            int ch = idx >> 5, l = idx & 31;
            int pos = kc * 32 + l;
            float val = 0.f;
            if (pos < NKTOT) {
                int gy = y0 - HALO + pos / WINS;
                int gx = x0 - HALO + pos % WINS;
                if (gy >= 0 && gy < HH && gx >= 0 && gx < WW)
                    val = vb[(size_t)ch * HW + gy * WW + gx];
            }
            vlds[ch * 34 + l] = f2bf(val);
        }
        __syncthreads();
        v16bf af = load_a_frag(attnlds + h * 2560, 160, kc * 32, lane);
#pragma unroll
        for (int nt = 0; nt < 4; ++nt) {
            v16bf vf = load_b_frag(vlds, 34, h * HD + nt * 16, 0, lane);
            o[nt] = wmma_bf16(af, vf, o[nt]);
        }
        __syncthreads();
    }

    const float r1 = rate1p[0];
#pragma unroll
    for (int nt = 0; nt < 4; ++nt) {
        int d = h * HD + nt * 16 + Nl;
        float* op = out + ((size_t)b * CCH + d) * HW;
#pragma unroll
        for (int r = 0; r < 8; ++r) {
            int m = mbase + r;
            op[(y0 + (m >> 2)) * WW + x0 + (m & 3)] = r1 * o[nt][r];
        }
    }
}

// =======================================================================
// Kernel 3: f2[b][j][m][p] = sum_c fc_w[j,c] * {q,k,v}[b, (c%4)*64+m, p]
// =======================================================================
__global__ __launch_bounds__(256) void proj_f2(
    const float* __restrict__ q, const float* __restrict__ k,
    const float* __restrict__ v, const float* __restrict__ fcw,
    float* __restrict__ f2) {
    int gid = blockIdx.x * 256 + threadIdx.x;
    int p = gid % HW;
    int m = (gid / HW) % HD;
    int j = (gid / (HW * HD)) % 9;
    int b = gid / (HW * HD * 9);
    float acc = 0.f;
    const size_t pb = (size_t)b * CCH * HW + p;
#pragma unroll
    for (int c4 = 0; c4 < 4; ++c4) {
        size_t off = (size_t)(c4 * HD + m) * HW;
        acc += q[pb + off] * fcw[j * 12 + c4];
        acc += k[pb + off] * fcw[j * 12 + 4 + c4];
        acc += v[pb + off] * fcw[j * 12 + 8 + c4];
    }
    f2[gid] = acc;
}

// =======================================================================
// Kernel 4: out += rate2 * depthwise3x3(f2)   (group g = o>>2, 9 in-ch per group)
// =======================================================================
__global__ __launch_bounds__(256) void dwconv_add(
    const float* __restrict__ f2, const float* __restrict__ depw,
    const float* __restrict__ rate2p, float* __restrict__ out) {
    int gid = blockIdx.x * 256 + threadIdx.x;
    int p = gid % HW;
    int o = (gid / HW) % CCH;
    int b = gid / (HW * CCH);
    int y = p / WW, x = p % WW;
    int g = o >> 2;
    float acc = 0.f;
#pragma unroll
    for (int j = 0; j < 9; ++j) {
        const float* fp = f2 + (((size_t)b * 9 + j) * HD + g) * HW;
        const float* wp = depw + o * 81 + j * 9;
#pragma unroll
        for (int ky = 0; ky < 3; ++ky) {
            int yy = y + ky - 1;
            if (yy < 0 || yy >= HH) continue;
#pragma unroll
            for (int kx = 0; kx < 3; ++kx) {
                int xx = x + kx - 1;
                if (xx < 0 || xx >= WW) continue;
                acc += fp[yy * WW + xx] * wp[ky * 3 + kx];
            }
        }
    }
    out[gid] += rate2p[0] * acc;
}

// =======================================================================
extern "C" void kernel_launch(void* const* d_in, const int* in_sizes, int n_in,
                              void* d_out, int out_size, void* d_ws, size_t ws_size,
                              hipStream_t stream) {
    const float* x     = (const float*)d_in[0];
    const float* w1    = (const float*)d_in[1];
    const float* b1    = (const float*)d_in[2];
    const float* w2    = (const float*)d_in[3];
    const float* b2    = (const float*)d_in[4];
    const float* w3    = (const float*)d_in[5];
    const float* b3    = (const float*)d_in[6];
    const float* fcw   = (const float*)d_in[7];
    const float* depw  = (const float*)d_in[8];
    const float* relh  = (const float*)d_in[9];
    const float* relw  = (const float*)d_in[10];
    const float* rate1 = (const float*)d_in[11];
    const float* rate2 = (const float*)d_in[12];
    float* out = (float*)d_out;

    const size_t QS  = (size_t)BSZ * CCH * HW;        // 12.8M elems
    const size_t F2S = (size_t)BSZ * 9 * HD * HW;     // 28.9M elems
    float* q  = (float*)d_ws;
    float* k  = q + QS;
    float* v  = k + QS;
    float* f2 = v + QS;
    unsigned short* xT  = (unsigned short*)(f2 + F2S);      // bf16 [b][p][c]
    unsigned short* wbf = xT + QS;                          // bf16 [3][o][c]

    x_to_bf16T<<<(int)(QS / 256), 256, 0, stream>>>(x, xT);
    w_to_bf16<<<(3 * CCH * CCH) / 256, 256, 0, stream>>>(w1, w2, w3, wbf);
    qkv_gemm<<<dim3(BSZ, 49, 2), 256, 0, stream>>>(xT, wbf, b1, b2, b3, q, k, v);
    halo_attn<<<dim3(BSZ, NWIN), 128, 0, stream>>>(q, k, v, relh, relw, rate1, out);
    proj_f2<<<(int)(F2S / 256), 256, 0, stream>>>(q, k, v, fcw, f2);
    dwconv_add<<<(int)(QS / 256), 256, 0, stream>>>(f2, depw, rate2, out);
}
